// Model_62766652063955
// MI455X (gfx1250) — compile-verified, matrix operand order checked by reading, not run on hardware
//
#include <hip/hip_runtime.h>
#include <hip/hip_fp16.h>

// ---------------- problem constants ----------------
#define B_    2
#define S_    4096
#define H_    12
#define D_    64
#define HID_  768
#define W_    256
#define FFN_  3072
#define NB_   16            // S / W
#define MR_   (B_ * S_)     // 8192 token rows

typedef __attribute__((ext_vector_type(16))) _Float16 v16h;
typedef __attribute__((ext_vector_type(8)))  _Float16 v8h;
typedef __attribute__((ext_vector_type(8)))  float    v8f;
typedef int v4i_ __attribute__((vector_size(16)));    // matches builtin pointee

typedef __attribute__((address_space(1))) v4i_* as1_v4i_p;  // global 16B chunk
typedef __attribute__((address_space(3))) v4i_* as3_v4i_p;  // LDS 16B chunk

// ---------------- CDNA5 async LDS staging ----------------
// GLOBAL_LOAD_ASYNC_TO_LDS_B128: per-lane 16B global -> LDS, tracked by ASYNCcnt.
// Generic->AS1 keeps the 64-bit VA; generic->AS3 truncates to addr[31:0],
// which per the ISA flat-aperture rules is exactly the LDS byte offset.
__device__ __forceinline__ void cp16_async(const _Float16* g, _Float16* l) {
#if __has_builtin(__builtin_amdgcn_global_load_async_to_lds_b128)
  __builtin_amdgcn_global_load_async_to_lds_b128(
      (as1_v4i_p)(uintptr_t)(const void*)g,
      (as3_v4i_p)(unsigned int)(uintptr_t)(void*)l,
      0, 0);
#else
  unsigned int loff = (unsigned int)(uintptr_t)(void*)l;  // LDS byte offset
  asm volatile("global_load_async_to_lds_b128 %0, %1, off"
               :: "v"(loff), "v"(g) : "memory");
#endif
}

__device__ __forceinline__ void wait_async_stage() {
#if __has_builtin(__builtin_amdgcn_s_wait_asynccnt)
  __builtin_amdgcn_s_wait_asynccnt(0);
#else
  asm volatile("s_wait_asynccnt 0x0" ::: "memory");
#endif
}

// ---------------- WMMA helpers ----------------
__device__ __forceinline__ v8f wmma16(v16h a, v16h b, v8f c) {
  // D = A(16x32 f16) * B(32x16 f16) + C(16x16 f32)
  return __builtin_amdgcn_wmma_f32_16x16x32_f16(
      /*neg_a=*/false, a, /*neg_b=*/false, b,
      /*c_mod=*/(short)0, c, /*reuse_a=*/false, /*reuse_b=*/false);
}

// Load a 16x32 f16 A-fragment from row-major storage (ld = row stride, halfs).
// Per ISA 7.12.2: lane L holds row M=L%16, K = {8*(L/16)+0..7, 16+8*(L/16)+0..7}.
// B operand uses the A layout applied to B^T, so B tiles are kept transposed
// (N-major rows of length K) and use this same loader.
__device__ __forceinline__ v16h ld_frag(const _Float16* p, int ld) {
  const int lane = threadIdx.x & 31;
  const int m    = lane & 15;
  const int kb   = (lane >> 4) << 3;      // 0 or 8
  const _Float16* r = p + m * ld + kb;
  v8h lo = *(const v8h*)(r);
  v8h hi = *(const v8h*)(r + 16);
  v16h a;
#pragma unroll
  for (int i = 0; i < 8; ++i) { a[i] = lo[i]; a[8 + i] = hi[i]; }
  return a;
}

__device__ __forceinline__ float gelu_t(float x) {
  float x3 = x * x * x;
  return 0.5f * x * (1.0f + tanhf(0.7978845608028654f * (x + 0.044715f * x3)));
}

// ---------------- small prep kernels ----------------
// f32 (K,N) weight -> f16 transposed (N,K), row-major
__global__ __launch_bounds__(256) void cvt_wT(const float* __restrict__ in,
                                              _Float16* __restrict__ out,
                                              int K, int N) {
  int tid = blockIdx.x * blockDim.x + threadIdx.x;
  if (tid >= N * K) return;
  int n = tid / K, k = tid - n * K;
  out[tid] = (_Float16)in[(size_t)k * N + n];
}

// x = tok_emb[ids] + pos_emb  (pre-LN)
__global__ __launch_bounds__(256) void embed_gather(const int* __restrict__ ids,
                                                    const float* __restrict__ tok,
                                                    const float* __restrict__ pos,
                                                    float* __restrict__ out) {
  int tid = blockIdx.x * blockDim.x + threadIdx.x;   // grid sized exactly
  int row = tid / HID_;                              // b*S + s
  int c   = tid - row * HID_;
  int s   = row & (S_ - 1);
  out[tid] = tok[(size_t)ids[row] * HID_ + c] + pos[(size_t)s * HID_ + c];
}

// out = LN(x [+ y]); also optional f16 copy.  One 256-thread block per row.
__global__ __launch_bounds__(256) void add_ln(const float* __restrict__ xin,
                                              const float* __restrict__ yin,
                                              const float* __restrict__ g,
                                              const float* __restrict__ bta,
                                              float* __restrict__ xout,
                                              _Float16* __restrict__ hout,
                                              int hasY) {
  const int row = blockIdx.x;
  const int tid = threadIdx.x;
  const float* xr = xin + (size_t)row * HID_;
  const float* yr = yin + (size_t)row * HID_;
  float v[3];
  float s = 0.f, s2 = 0.f;
#pragma unroll
  for (int i = 0; i < 3; ++i) {
    int c = tid + i * 256;
    float t = xr[c] + (hasY ? yr[c] : 0.f);
    v[i] = t; s += t; s2 += t * t;
  }
  __shared__ float rs[256], rq[256];
  rs[tid] = s; rq[tid] = s2;
  __syncthreads();
  for (int off = 128; off > 0; off >>= 1) {
    if (tid < off) { rs[tid] += rs[tid + off]; rq[tid] += rq[tid + off]; }
    __syncthreads();
  }
  float mean = rs[0] * (1.0f / HID_);
  float var  = rq[0] * (1.0f / HID_) - mean * mean;
  float inv  = rsqrtf(var + 1e-5f);
#pragma unroll
  for (int i = 0; i < 3; ++i) {
    int c = tid + i * 256;
    float o = (v[i] - mean) * inv * g[c] + bta[c];
    xout[(size_t)row * HID_ + c] = o;
    if (hout) hout[(size_t)row * HID_ + c] = (_Float16)o;
  }
}

// ---------------- tiled WMMA GEMM ----------------
// C[M,N] = act( (A[M,K] @ Bt[N,K]^T + bias[N]) * alpha )
// Block tile 128x128, 8 waves (4x2), wave tile 32x64 (2x4 WMMA tiles), BK=32.
// Double-buffered LDS staged with GLOBAL_LOAD_ASYNC_TO_LDS_B128.
template <int ACT>
__global__ __launch_bounds__(256)
void gemm_f16(const _Float16* __restrict__ A, const _Float16* __restrict__ Bt,
              const float* __restrict__ bias, float alpha,
              float* __restrict__ Cf, _Float16* __restrict__ Ch,
              int M, int N, int K) {
  constexpr int BM = 128, BN = 128, BK = 32;
  __shared__ __align__(16) _Float16 sA[2][BM * BK];
  __shared__ __align__(16) _Float16 sB[2][BN * BK];
  const int tid  = threadIdx.x;
  const int wv   = tid >> 5;
  const int wm   = wv & 3, wn = wv >> 2;   // 4x2 wave grid
  const int lane = tid & 31, hlf = lane >> 4, l15 = lane & 15;
  const int m0 = blockIdx.y * BM;
  const int n0 = blockIdx.x * BN;

  v8f zero = {};
  v8f acc[2][4];
#pragma unroll
  for (int mi = 0; mi < 2; ++mi)
#pragma unroll
    for (int ni = 0; ni < 4; ++ni) acc[mi][ni] = zero;

  auto stage = [&](int buf, int k0) {
    for (int c = tid; c < BM * 4; c += 256) {          // A: 512 x 16B chunks
      int r = c >> 2, co = (c & 3) << 3;
      cp16_async(&A[(size_t)(m0 + r) * K + k0 + co], &sA[buf][r * BK + co]);
    }
    for (int c = tid; c < BN * 4; c += 256) {          // Bt: 512 x 16B chunks
      int r = c >> 2, co = (c & 3) << 3;
      cp16_async(&Bt[(size_t)(n0 + r) * K + k0 + co], &sB[buf][r * BK + co]);
    }
  };

  const int nstages = K / BK;
  stage(0, 0);
  wait_async_stage();
  __syncthreads();

  for (int s = 0; s < nstages; ++s) {
    const int cur = s & 1;
    // fragments for this stage (issued before staging the next tile)
    v16h aF0 = ld_frag(&sA[cur][(wm * 32) * BK], BK);
    v16h aF1 = ld_frag(&sA[cur][(wm * 32 + 16) * BK], BK);
    v16h bF[4];
#pragma unroll
    for (int ni = 0; ni < 4; ++ni)
      bF[ni] = ld_frag(&sB[cur][(wn * 64 + ni * 16) * BK], BK);

    if (s + 1 < nstages) stage(cur ^ 1, (s + 1) * BK);

#pragma unroll
    for (int ni = 0; ni < 4; ++ni) {
      acc[0][ni] = wmma16(aF0, bF[ni], acc[0][ni]);
      acc[1][ni] = wmma16(aF1, bF[ni], acc[1][ni]);
    }
    wait_async_stage();
    __syncthreads();
  }

#pragma unroll
  for (int mi = 0; mi < 2; ++mi)
#pragma unroll
    for (int ni = 0; ni < 4; ++ni)
#pragma unroll
      for (int v = 0; v < 8; ++v) {
        int row = m0 + wm * 32 + mi * 16 + v + 8 * hlf;
        int col = n0 + wn * 64 + ni * 16 + l15;
        float t = (acc[mi][ni][v] + bias[col]) * alpha;
        if (ACT == 1) t = gelu_t(t);
        if (Cf) Cf[(size_t)row * N + col] = t;
        if (Ch) Ch[(size_t)row * N + col] = (_Float16)t;
      }
}

// ---------------- banded flash attention ----------------
// One block = (b, h, nb, 64-query subtile). 4 waves, wave w owns 16 query rows.
// Keys streamed in 64-wide chunks over the 3W=768 band window; online softmax.
__global__ __launch_bounds__(128)
void attn_band(const _Float16* __restrict__ Q, const _Float16* __restrict__ Km,
               const _Float16* __restrict__ Vm, const int* __restrict__ amask,
               _Float16* __restrict__ CTX) {
  int t = blockIdx.x;
  const int qsub = t & 3;  t >>= 2;
  const int nb   = t & 15; t >>= 4;
  const int h    = t % H_;
  const int b    = t / H_;

  const int tid  = threadIdx.x;
  const int wv   = tid >> 5;
  const int lane = tid & 31, hlf = lane >> 4, l15 = lane & 15;

  __shared__ __align__(16) _Float16 sQ[64 * 64];
  __shared__ __align__(16) _Float16 sK[64 * 64];
  __shared__ __align__(16) _Float16 sVt[64 * 64];   // transposed: [d][key]
  __shared__ __align__(16) _Float16 sP[4][16 * 64]; // per-wave P tile
  __shared__ float sMask[64];                       // per-key additive mask

  // stage Q tile (64x64) asynchronously: 512 chunks / 128 threads
  for (int c = tid; c < 512; c += 128) {
    int r = c >> 3, co = (c & 7) << 3;
    int sq = nb * W_ + qsub * 64 + r;
    cp16_async(&Q[((size_t)(b * S_ + sq)) * HID_ + h * 64 + co],
               &sQ[r * 64 + co]);
  }
  wait_async_stage();
  __syncthreads();

  v16h aQ0 = ld_frag(&sQ[(wv * 16) * 64], 64);
  v16h aQ1 = ld_frag(&sQ[(wv * 16) * 64 + 32], 64);

  v8f zero = {};
  v8f accO[4]; accO[0] = zero; accO[1] = zero; accO[2] = zero; accO[3] = zero;
  float m8[8], l8[8];
#pragma unroll
  for (int v = 0; v < 8; ++v) { m8[v] = -3.0e38f; l8[v] = 0.f; }

  for (int kc = 0; kc < 12; ++kc) {          // 12 * 64 = 768 keys in window
    __syncthreads();
    // stage K chunk (async, row-major) and V chunk (sync, transposed into sVt)
    for (int c = tid; c < 512; c += 128) {
      int r = c >> 3, co = (c & 7) << 3;
      int ak = nb * W_ - W_ + kc * 64 + r;   // absolute key index
      v8h vv = {};
      if (ak >= 0 && ak < S_) {
        size_t base = ((size_t)(b * S_ + ak)) * HID_ + h * 64 + co;
        cp16_async(&Km[base], &sK[r * 64 + co]);
        vv = *(const v8h*)&Vm[base];
      } else {
        *(v8h*)&sK[r * 64 + co] = vv;        // zero-fill out-of-range rows
      }
#pragma unroll
      for (int i = 0; i < 8; ++i) sVt[(co + i) * 64 + r] = vv[i];
    }
    if (tid < 64) {
      int ak = nb * W_ - W_ + kc * 64 + tid;
      float mv = -1.0e9f;
      if (ak >= 0 && ak < S_ && amask[b * S_ + ak] != 0) mv = 0.f;
      sMask[tid] = mv;
    }
    wait_async_stage();
    __syncthreads();

    // scores: 16 rows x 64 keys per wave (4 N-subtiles)
    v8f sc[4];
#pragma unroll
    for (int nt = 0; nt < 4; ++nt) {
      v8f c = zero;
      c = wmma16(aQ0, ld_frag(&sK[(nt * 16) * 64], 64), c);
      c = wmma16(aQ1, ld_frag(&sK[(nt * 16) * 64 + 32], 64), c);
      sc[nt] = c;
    }

    // mask + online softmax (per row = (v, lane-half))
#pragma unroll
    for (int v = 0; v < 8; ++v) {
      int r  = wv * 16 + v + 8 * hlf;        // local q row (0..63)
      int qi = qsub * 64 + r;                // row in block (0..255)
      float mrow = -3.0e38f;
#pragma unroll
      for (int nt = 0; nt < 4; ++nt) {
        int kj = kc * 64 + nt * 16 + l15;    // key index in window (0..767)
        bool band = (kj >= qi) && (kj <= qi + 2 * W_);
        float s = band ? (sc[nt][v] + sMask[nt * 16 + l15]) : -1.0e9f;
        sc[nt][v] = s;
        mrow = fmaxf(mrow, s);
      }
      mrow = fmaxf(mrow, __shfl_xor(mrow, 1, 32));
      mrow = fmaxf(mrow, __shfl_xor(mrow, 2, 32));
      mrow = fmaxf(mrow, __shfl_xor(mrow, 4, 32));
      mrow = fmaxf(mrow, __shfl_xor(mrow, 8, 32));
      float mnew  = fmaxf(m8[v], mrow);
      float scale = __expf(m8[v] - mnew);
      m8[v] = mnew;
      l8[v] *= scale;
#pragma unroll
      for (int nt = 0; nt < 4; ++nt) accO[nt][v] *= scale;
      float rsum = 0.f;
#pragma unroll
      for (int nt = 0; nt < 4; ++nt) {
        float p = __expf(sc[nt][v] - mnew);
        sc[nt][v] = p;
        rsum += p;
      }
      rsum += __shfl_xor(rsum, 1, 32);
      rsum += __shfl_xor(rsum, 2, 32);
      rsum += __shfl_xor(rsum, 4, 32);
      rsum += __shfl_xor(rsum, 8, 32);
      l8[v] += rsum;
#pragma unroll
      for (int nt = 0; nt < 4; ++nt)
        sP[wv][(v + 8 * hlf) * 64 + nt * 16 + l15] = (_Float16)sc[nt][v];
    }

    // ctx += P(16x64) @ V(64x64)   (intra-wave LDS round trip for P)
    v16h pa0 = ld_frag(&sP[wv][0], 64);
    v16h pa1 = ld_frag(&sP[wv][32], 64);
#pragma unroll
    for (int nt = 0; nt < 4; ++nt) {
      accO[nt] = wmma16(pa0, ld_frag(&sVt[(nt * 16) * 64], 64), accO[nt]);
      accO[nt] = wmma16(pa1, ld_frag(&sVt[(nt * 16) * 64 + 32], 64), accO[nt]);
    }
  }

  // write ctx in (B, S, H*D) layout, f16 for the O-projection GEMM
#pragma unroll
  for (int nt = 0; nt < 4; ++nt)
#pragma unroll
    for (int v = 0; v < 8; ++v) {
      int r  = wv * 16 + v + 8 * hlf;
      int sq = nb * W_ + qsub * 64 + r;
      int d  = nt * 16 + l15;
      float o = accO[nt][v] / l8[v];
      CTX[((size_t)(b * S_ + sq)) * HID_ + h * 64 + d] = (_Float16)o;
    }
}

// ---------------- pooler + classifier ----------------
__global__ __launch_bounds__(256)
void pool_cls(const float* __restrict__ X3, const float* __restrict__ pw,
              const float* __restrict__ pb, const float* __restrict__ cw,
              const float* __restrict__ cb, float* __restrict__ out) {
  const int b = blockIdx.x;
  const float* x = X3 + (size_t)b * S_ * HID_;   // row 0 of batch b
  __shared__ float pooled[HID_];
  for (int j = threadIdx.x; j < HID_; j += blockDim.x) {
    float acc = pb[j];
    for (int k = 0; k < HID_; ++k) acc += x[k] * pw[(size_t)k * HID_ + j];
    pooled[j] = tanhf(acc);
  }
  __syncthreads();
  if (threadIdx.x < 2) {
    int c = threadIdx.x;
    float acc = cb[c];
    for (int k = 0; k < HID_; ++k) acc += pooled[k] * cw[(size_t)k * 2 + c];
    out[b * 2 + c] = acc;
  }
}

// ---------------- workspace layout (bytes) ----------------
constexpr size_t SZ_XF  = (size_t)MR_ * HID_ * 4;   // f32 activation
constexpr size_t SZ_XH  = (size_t)MR_ * HID_ * 2;   // f16 activation
constexpr size_t SZ_HH  = (size_t)MR_ * FFN_ * 2;   // f16 FFN hidden
constexpr size_t SZ_WT  = (size_t)HID_ * HID_ * 2;
constexpr size_t SZ_W1T = (size_t)HID_ * FFN_ * 2;

constexpr size_t OFF_X   = 0;
constexpr size_t OFF_X2  = OFF_X   + SZ_XF;
constexpr size_t OFF_X3  = OFF_X2  + SZ_XF;
constexpr size_t OFF_TMP = OFF_X3  + SZ_XF;
constexpr size_t OFF_XH  = OFF_TMP + SZ_XF;   // also reused as CTXH
constexpr size_t OFF_X2H = OFF_XH  + SZ_XH;
constexpr size_t OFF_QH  = OFF_X2H + SZ_XH;
constexpr size_t OFF_KH  = OFF_QH  + SZ_XH;
constexpr size_t OFF_VH  = OFF_KH  + SZ_XH;
constexpr size_t OFF_HH  = OFF_VH  + SZ_XH;
constexpr size_t OFF_WQT = OFF_HH  + SZ_HH;
constexpr size_t OFF_WKT = OFF_WQT + SZ_WT;
constexpr size_t OFF_WVT = OFF_WKT + SZ_WT;
constexpr size_t OFF_WOT = OFF_WVT + SZ_WT;
constexpr size_t OFF_W1T = OFF_WOT + SZ_WT;
constexpr size_t OFF_W2T = OFF_W1T + SZ_W1T;

extern "C" void kernel_launch(void* const* d_in, const int* in_sizes, int n_in,
                              void* d_out, int out_size, void* d_ws, size_t ws_size,
                              hipStream_t stream) {
  (void)in_sizes; (void)n_in; (void)out_size; (void)ws_size;
  const int*   ids   = (const int*)d_in[0];
  const int*   amask = (const int*)d_in[1];
  const float* tok   = (const float*)d_in[2];
  const float* pos   = (const float*)d_in[3];
  const float* eg    = (const float*)d_in[4];
  const float* eb    = (const float*)d_in[5];
  const float* wq = (const float*)d_in[6],  *bq = (const float*)d_in[7];
  const float* wk = (const float*)d_in[8],  *bk = (const float*)d_in[9];
  const float* wv = (const float*)d_in[10], *bv = (const float*)d_in[11];
  const float* wo = (const float*)d_in[12], *bo = (const float*)d_in[13];
  const float* ag = (const float*)d_in[14], *ab = (const float*)d_in[15];
  const float* w1 = (const float*)d_in[16], *b1 = (const float*)d_in[17];
  const float* w2 = (const float*)d_in[18], *b2 = (const float*)d_in[19];
  const float* fg = (const float*)d_in[20], *fb = (const float*)d_in[21];
  const float* pw = (const float*)d_in[22], *pb = (const float*)d_in[23];
  const float* cw = (const float*)d_in[24], *cb = (const float*)d_in[25];

  char* ws = (char*)d_ws;
  float*    X    = (float*)(ws + OFF_X);
  float*    X2   = (float*)(ws + OFF_X2);
  float*    X3   = (float*)(ws + OFF_X3);
  float*    TMP  = (float*)(ws + OFF_TMP);
  _Float16* XH   = (_Float16*)(ws + OFF_XH);
  _Float16* CTXH = (_Float16*)(ws + OFF_XH);   // alias: XH dead after QKV GEMMs
  _Float16* X2H  = (_Float16*)(ws + OFF_X2H);
  _Float16* QH   = (_Float16*)(ws + OFF_QH);
  _Float16* KH   = (_Float16*)(ws + OFF_KH);
  _Float16* VH   = (_Float16*)(ws + OFF_VH);
  _Float16* HH   = (_Float16*)(ws + OFF_HH);
  _Float16* WQT  = (_Float16*)(ws + OFF_WQT);
  _Float16* WKT  = (_Float16*)(ws + OFF_WKT);
  _Float16* WVT  = (_Float16*)(ws + OFF_WVT);
  _Float16* WOT  = (_Float16*)(ws + OFF_WOT);
  _Float16* W1T  = (_Float16*)(ws + OFF_W1T);
  _Float16* W2T  = (_Float16*)(ws + OFF_W2T);

  // --- weight conversion (f32 (K,N) -> f16 (N,K)) ---
  cvt_wT<<<(HID_ * HID_) / 256, 256, 0, stream>>>(wq, WQT, HID_, HID_);
  cvt_wT<<<(HID_ * HID_) / 256, 256, 0, stream>>>(wk, WKT, HID_, HID_);
  cvt_wT<<<(HID_ * HID_) / 256, 256, 0, stream>>>(wv, WVT, HID_, HID_);
  cvt_wT<<<(HID_ * HID_) / 256, 256, 0, stream>>>(wo, WOT, HID_, HID_);
  cvt_wT<<<(HID_ * FFN_) / 256, 256, 0, stream>>>(w1, W1T, HID_, FFN_);
  cvt_wT<<<(FFN_ * HID_) / 256, 256, 0, stream>>>(w2, W2T, FFN_, HID_);

  // --- embedding + LN ---
  embed_gather<<<(MR_ * HID_) / 256, 256, 0, stream>>>(ids, tok, pos, TMP);
  add_ln<<<MR_, 256, 0, stream>>>(TMP, TMP, eg, eb, X, XH, 0);

  // --- QKV projections (q scaled by 1/sqrt(D)) ---
  dim3 gP(HID_ / 128, MR_ / 128);
  gemm_f16<0><<<gP, 256, 0, stream>>>(XH, WQT, bq, 0.125f, nullptr, QH, MR_, HID_, HID_);
  gemm_f16<0><<<gP, 256, 0, stream>>>(XH, WKT, bk, 1.0f,   nullptr, KH, MR_, HID_, HID_);
  gemm_f16<0><<<gP, 256, 0, stream>>>(XH, WVT, bv, 1.0f,   nullptr, VH, MR_, HID_, HID_);

  // --- banded attention ---
  attn_band<<<B_ * H_ * NB_ * 4, 128, 0, stream>>>(QH, KH, VH, amask, CTXH);

  // --- output projection + residual LN ---
  gemm_f16<0><<<gP, 256, 0, stream>>>(CTXH, WOT, bo, 1.0f, TMP, nullptr, MR_, HID_, HID_);
  add_ln<<<MR_, 256, 0, stream>>>(X, TMP, ag, ab, X2, X2H, 1);

  // --- FFN ---
  dim3 gF1(FFN_ / 128, MR_ / 128);
  gemm_f16<1><<<gF1, 256, 0, stream>>>(X2H, W1T, b1, 1.0f, nullptr, HH, MR_, FFN_, HID_);
  gemm_f16<0><<<gP,  256, 0, stream>>>(HH,  W2T, b2, 1.0f, TMP, nullptr, MR_, HID_, FFN_);
  add_ln<<<MR_, 256, 0, stream>>>(X2, TMP, fg, fb, X3, nullptr, 1);

  // --- pooler + classifier ---
  pool_cls<<<B_, 256, 0, stream>>>(X3, pw, pb, cw, cb, (float*)d_out);
}
